// GPT2Attention_74594991997083
// MI455X (gfx1250) — matrix operand order repaired and checked by reading, not compiled, not run on hardware
//
#include <hip/hip_runtime.h>

#define DEVI __device__ __forceinline__

typedef __attribute__((ext_vector_type(16))) __bf16        v16bf;
typedef __attribute__((ext_vector_type(8)))  float         v8f;
typedef __attribute__((ext_vector_type(4)))  float         f32x4;
typedef __attribute__((ext_vector_type(4)))  unsigned int  u32x4;
typedef __attribute__((ext_vector_type(2)))  unsigned int  u32x2;
typedef __attribute__((ext_vector_type(4)))  int           i32x4;

static constexpr int BB = 2, TT = 2048, CC = 1024, HH = 16, DD = 64;
static constexpr int MROWS = BB * TT;  // 4096

union ABfrag {
  __bf16 u[16];
  u32x4  q[2];
  v16bf  v;
};

DEVI v8f wmma_bf16(v16bf a, v16bf b, v8f c) {
  return __builtin_amdgcn_wmma_f32_16x16x32_bf16(
      /*neg_a=*/false, a, /*neg_b=*/false, b,
      /*c_mod=*/(short)0, c, /*reuse_a=*/false, /*reuse_b=*/false);
}

// ---- CDNA5 async global->LDS staging (ASYNCcnt tracked), guarded ----------
#if __has_builtin(__builtin_amdgcn_global_load_async_to_lds_b128)
#define HAVE_ASYNC 1
DEVI void async_cp16B(const __bf16* g, __bf16* l) {
  __builtin_amdgcn_global_load_async_to_lds_b128(
      (__attribute__((address_space(1))) i32x4*)g,
      (__attribute__((address_space(3))) i32x4*)l, 0, 0);
}
DEVI void wait_async0() {
#if __has_builtin(__builtin_amdgcn_s_wait_asynccnt)
  __builtin_amdgcn_s_wait_asynccnt(0);
#else
  asm volatile("s_wait_asynccnt 0" ::: "memory");
#endif
}
#else
#define HAVE_ASYNC 0
#endif

// ---------------------------------------------------------------------------
// fp32 -> bf16 elementwise (4 elems/thread, vectorized)
// ---------------------------------------------------------------------------
__global__ __launch_bounds__(256) void f32_to_bf16_kernel(
    const float* __restrict__ src, __bf16* __restrict__ dst) {
  size_t i = ((size_t)blockIdx.x * 256 + threadIdx.x) * 4;
  f32x4 f = *(const f32x4*)(src + i);
  union { __bf16 h[4]; u32x2 v; } o;
  o.h[0] = (__bf16)f.x; o.h[1] = (__bf16)f.y;
  o.h[2] = (__bf16)f.z; o.h[3] = (__bf16)f.w;
  *(u32x2*)(dst + i) = o.v;
}

// ---------------------------------------------------------------------------
// fp32 [K,N] -> bf16 transposed [N,K], LDS-tiled (coalesced both sides)
// ---------------------------------------------------------------------------
__global__ __launch_bounds__(256) void transpose_conv_kernel(
    const float* __restrict__ W, __bf16* __restrict__ WT, int K, int N) {
  __shared__ __bf16 tile[16][17];
  const int k0 = blockIdx.y * 16, n0 = blockIdx.x * 16;
  const int r = threadIdx.x >> 4, c = threadIdx.x & 15;
  tile[r][c] = (__bf16)W[(size_t)(k0 + r) * N + (n0 + c)];
  __syncthreads();
  WT[(size_t)(n0 + r) * K + (k0 + c)] = tile[c][r];
}

// ---------------------------------------------------------------------------
// GEMM: Cout[M,N] = A[M,K] @ WT[N,K]^T + bias[N]   (bf16 in, fp32 out)
// Block: 256 thr (8 waves). Block tile 64x128, K-step 32. Wave -> 16x64 tile.
// A tile: async double-buffered LDS. B frags: direct 32B loads (L2-resident).
// ---------------------------------------------------------------------------
__global__ __launch_bounds__(256) void gemm_bf16_kernel(
    const __bf16* __restrict__ A, const __bf16* __restrict__ WT,
    const float* __restrict__ bias, float* __restrict__ Cout,
    int M, int N, int K) {
  __shared__ __bf16 As[2][64][48];  // 32 data cols, 96B row stride (16B mult)

  const int tid  = threadIdx.x;
  const int lane = tid & 31, wave = tid >> 5;
  const int half = lane >> 4, ln = lane & 15;
  const int wr = wave & 3, wc = wave >> 2;
  const int m0 = blockIdx.y * 64, n0 = blockIdx.x * 128;

  v8f acc[4] = {v8f{}, v8f{}, v8f{}, v8f{}};

  const int ar = tid >> 2, ac = (tid & 3) * 8;  // A stage coords (64x32)
  const __bf16* aprow = A + (size_t)(m0 + ar) * K + ac;

  // prologue: stage k0 = 0 into buffer 0
#if HAVE_ASYNC
  async_cp16B(aprow, &As[0][ar][ac]);
#else
  *(u32x4*)&As[0][ar][ac] = *(const u32x4*)aprow;
#endif

  const int nsteps = K / 32;
  for (int i = 0; i < nsteps; ++i) {
    const int k0 = i * 32;
#if HAVE_ASYNC
    wait_async0();
#endif
    __syncthreads();  // stage i ready everywhere; stage i-1 fully consumed
    if (i + 1 < nsteps) {
#if HAVE_ASYNC
      async_cp16B(aprow + k0 + 32, &As[(i + 1) & 1][ar][ac]);
#else
      *(u32x4*)&As[(i + 1) & 1][ar][ac] = *(const u32x4*)(aprow + k0 + 32);
#endif
    }

    ABfrag a;                       // A: K = e + half*16
    const int arow = wr * 16 + ln;
    a.q[0] = *(const u32x4*)(&As[i & 1][arow][half * 16]);
    a.q[1] = *(const u32x4*)(&As[i & 1][arow][half * 16 + 8]);

    #pragma unroll
    for (int nt = 0; nt < 4; ++nt) {   // B: K = e + half*16 -> contiguous 32B
      const int col = n0 + wc * 64 + nt * 16 + ln;
      v16bf b = *(const v16bf*)(WT + (size_t)col * K + k0 + half * 16);
      acc[nt] = wmma_bf16(a.v, b, acc[nt]);
    }
  }

  #pragma unroll
  for (int nt = 0; nt < 4; ++nt) {
    #pragma unroll
    for (int v = 0; v < 8; ++v) {   // C/D: row = v + half*8, col = lane%16
      int row = m0 + wr * 16 + half * 8 + v;
      int col = n0 + wc * 64 + nt * 16 + ln;
      Cout[(size_t)row * N + col] = acc[nt][v] + bias[col];
    }
  }
}

// ---------------------------------------------------------------------------
// Repack qkv fp32 [B,T,3C] -> bf16 q[B,H,T,D] (x 1/sqrt(D)), k[B,H,T,D],
// vT[B,H,D,T]
// ---------------------------------------------------------------------------
__global__ __launch_bounds__(256) void qkv_scatter_kernel(
    const float* __restrict__ qkv, __bf16* __restrict__ qb,
    __bf16* __restrict__ kb, __bf16* __restrict__ vT) {
  size_t idx = (size_t)blockIdx.x * 256 + threadIdx.x;
  const int n3 = 3 * CC;
  int m = (int)(idx / n3);
  int n = (int)(idx % n3);
  int b = m / TT, t = m % TT;
  int sel = n / CC, c = n % CC;
  int h = c / DD, d = c % DD;
  float val = qkv[idx];
  size_t bh = (size_t)(b * HH + h);
  if (sel == 0)      qb[(bh * TT + t) * DD + d] = (__bf16)(val * 0.125f);  // 1/sqrt(64)
  else if (sel == 1) kb[(bh * TT + t) * DD + d] = (__bf16)val;
  else               vT[(bh * DD + d) * TT + t] = (__bf16)val;
}

// ---------------------------------------------------------------------------
// Flash attention (causal). Grid (B*H, T/128), 256 thr. Wave w owns query
// tile q0 = qblk*128 + w*16. Computes S^T = K*Q^T and Y^T = V^T * P^T so the
// softmax row (= query) maps to lane%16: stats are per-lane scalars.
// Output written directly as bf16 [B,T,C] for the out-proj GEMM.
// ---------------------------------------------------------------------------
__global__ __launch_bounds__(256) void attn_kernel(
    const __bf16* __restrict__ qb, const __bf16* __restrict__ kbuf,
    const __bf16* __restrict__ vT, __bf16* __restrict__ y) {
  const int bh   = blockIdx.x;
  const int qblk = blockIdx.y;
  const int lane = threadIdx.x & 31, wave = threadIdx.x >> 5;
  const int half = lane >> 4, qq = lane & 15;
  const int q0 = qblk * 128 + wave * 16;
  const int query = q0 + qq;

  const __bf16* qrow  = qb   + ((size_t)bh * TT + query) * DD;
  const __bf16* kbase = kbuf + (size_t)bh * TT * DD;
  const __bf16* vbase = vT   + (size_t)bh * DD * TT;

  // Q as B-matrix (N=query, K=d): contiguous 16 bf16, 32B aligned
  v16bf bq0 = *(const v16bf*)(qrow + half * 16);        // d = half*16 + e
  v16bf bq1 = *(const v16bf*)(qrow + 32 + half * 16);   // d = 32 + half*16 + e

  float m = -__builtin_inff();
  float l = 0.f;
  v8f acc[4] = {v8f{}, v8f{}, v8f{}, v8f{}};            // Y^T, dtile 0..3

  const int kmax = qblk * 128 + 128;                    // uniform per block
  for (int kb = 0; kb < kmax; kb += 32) {
    // ---- S^T = K * Q^T : two 16(key)x16(query) tiles, K-dim = d (64) ----
    v8f s0 = {}; v8f s1 = {};
    {
      const __bf16* kp = kbase + (size_t)(kb + qq) * DD;  // A row M=key
      ABfrag ak;
      ak.q[0] = *(const u32x4*)(kp + half * 16);
      ak.q[1] = *(const u32x4*)(kp + half * 16 + 8);
      s0 = wmma_bf16(ak.v, bq0, s0);
      ak.q[0] = *(const u32x4*)(kp + 32 + half * 16);
      ak.q[1] = *(const u32x4*)(kp + 40 + half * 16);
      s0 = wmma_bf16(ak.v, bq1, s0);
    }
    {
      const __bf16* kp = kbase + (size_t)(kb + 16 + qq) * DD;
      ABfrag ak;
      ak.q[0] = *(const u32x4*)(kp + half * 16);
      ak.q[1] = *(const u32x4*)(kp + half * 16 + 8);
      s1 = wmma_bf16(ak.v, bq0, s1);
      ak.q[0] = *(const u32x4*)(kp + 32 + half * 16);
      ak.q[1] = *(const u32x4*)(kp + 40 + half * 16);
      s1 = wmma_bf16(ak.v, bq1, s1);
    }

    // ---- causal mask + row max (row = query = lane%16) ----
    float rmax = -__builtin_inff();
    #pragma unroll
    for (int v = 0; v < 8; ++v) {          // C row = v + half*8 = key-in-tile
      int key0 = kb + half * 8 + v;
      int key1 = key0 + 16;
      if (key0 > query) s0[v] = -__builtin_inff();
      if (key1 > query) s1[v] = -__builtin_inff();
      rmax = fmaxf(rmax, fmaxf(s0[v], s1[v]));
    }
    rmax = fmaxf(rmax, __shfl_xor(rmax, 16, 32));   // combine lane-halves

    float mnew = fmaxf(m, rmax);
    float cf   = __expf(m - mnew);                  // exp(-inf)=0 on first iter
    float p0[8], p1[8];
    float psum = 0.f;
    #pragma unroll
    for (int v = 0; v < 8; ++v) {
      p0[v] = __expf(s0[v] - mnew);
      p1[v] = __expf(s1[v] - mnew);
      psum += p0[v] + p1[v];
    }
    psum += __shfl_xor(psum, 16, 32);
    l = l * cf + psum;
    m = mnew;
    #pragma unroll
    for (int dt = 0; dt < 4; ++dt)
      #pragma unroll
      for (int v = 0; v < 8; ++v) acc[dt][v] *= cf;

    // ---- P^T C-layout -> B-layout via shuffles ----
    // B elem e: key-in-32 = e + half*16 -> tile_{half}, key-in-tile = e.
    // Source of (key e, query qq): lane = qq + (e>>3)*16, reg index e&7.
    ABfrag pb;
    #pragma unroll
    for (int e = 0; e < 16; ++e) {
      int src = qq + ((e >> 3) << 4);
      float t0 = __shfl(p0[e & 7], src, 32);
      float t1 = __shfl(p1[e & 7], src, 32);
      pb.u[e] = (__bf16)(half ? t1 : t0);
    }

    // ---- Y^T += V^T * P^T : A = V^T (M=d, K=key), contiguous in vT ----
    #pragma unroll
    for (int dt = 0; dt < 4; ++dt) {
      const __bf16* vp = vbase + (size_t)(dt * 16 + qq) * TT + kb;
      ABfrag av;
      av.q[0] = *(const u32x4*)(vp + half * 16);
      av.q[1] = *(const u32x4*)(vp + half * 16 + 8);
      acc[dt] = wmma_bf16(av.v, pb.v, acc[dt]);
    }
  }

  // ---- epilogue: y[b, t, h*D + d] = Y^T[d][query] / l  (bf16) ----
  float invl = 1.f / l;
  int b = bh / HH, h = bh % HH;
  __bf16* yrow = y + ((size_t)(b * TT + query)) * CC + h * DD;
  #pragma unroll
  for (int dt = 0; dt < 4; ++dt)
    #pragma unroll
    for (int v = 0; v < 8; ++v)
      yrow[dt * 16 + half * 8 + v] = (__bf16)(acc[dt][v] * invl);
}

// ---------------------------------------------------------------------------
extern "C" void kernel_launch(void* const* d_in, const int* in_sizes, int n_in,
                              void* d_out, int out_size, void* d_ws, size_t ws_size,
                              hipStream_t stream) {
  (void)in_sizes; (void)n_in; (void)out_size; (void)ws_size;
  const float* x      = (const float*)d_in[0];
  const float* W_attn = (const float*)d_in[1];
  const float* b_attn = (const float*)d_in[2];
  const float* W_proj = (const float*)d_in[3];
  const float* b_proj = (const float*)d_in[4];
  float* out = (float*)d_out;

  char* ws = (char*)d_ws;
  __bf16* xb  = (__bf16*)ws;  ws += (size_t)MROWS * CC * 2;          // 8 MB
  __bf16* WaT = (__bf16*)ws;  ws += (size_t)CC * 3 * CC * 2;         // 6.3 MB  [3C,C]
  __bf16* WpT = (__bf16*)ws;  ws += (size_t)CC * CC * 2;             // 2.1 MB  [C,C]
  float*  qkv = (float*)ws;   ws += (size_t)MROWS * 3 * CC * 4;      // 50.3 MB
  __bf16* qb  = (__bf16*)ws;  ws += (size_t)BB * HH * TT * DD * 2;   // 8 MB
  __bf16* kb  = (__bf16*)ws;  ws += (size_t)BB * HH * TT * DD * 2;   // 8 MB
  __bf16* vT  = (__bf16*)ws;  ws += (size_t)BB * HH * TT * DD * 2;   // 8 MB
  __bf16* ybf = (__bf16*)ws;                                         // 8 MB

  // 0) per-call fp32->bf16 conversions; weights transposed to [N,K]
  f32_to_bf16_kernel<<<dim3((unsigned)((size_t)MROWS * CC / 1024)), 256, 0, stream>>>(x, xb);
  transpose_conv_kernel<<<dim3((3 * CC) / 16, CC / 16), 256, 0, stream>>>(W_attn, WaT, CC, 3 * CC);
  transpose_conv_kernel<<<dim3(CC / 16, CC / 16), 256, 0, stream>>>(W_proj, WpT, CC, CC);

  // 1) qkv = x @ W_attn + b_attn
  gemm_bf16_kernel<<<dim3((3 * CC) / 128, MROWS / 64), 256, 0, stream>>>(
      xb, WaT, b_attn, qkv, MROWS, 3 * CC, CC);

  // 2) repack to per-head bf16 layouts (q pre-scaled by 1/sqrt(D))
  size_t total = (size_t)MROWS * 3 * CC;
  qkv_scatter_kernel<<<dim3((unsigned)(total / 256)), 256, 0, stream>>>(qkv, qb, kb, vT);

  // 3) fused causal flash attention -> ybf [B,T,C] bf16
  attn_kernel<<<dim3(BB * HH, TT / 128), 256, 0, stream>>>(qb, kb, vT, ybf);

  // 4) out = y @ W_proj + b_proj
  gemm_bf16_kernel<<<dim3(CC / 128, MROWS / 64), 256, 0, stream>>>(
      ybf, WpT, b_proj, out, MROWS, CC, CC);
}
